// SimpleSNN_66494683676947
// MI455X (gfx1250) — compile-verified, matrix operand order checked by reading, not run on hardware
//
#include <hip/hip_runtime.h>

typedef _Float16 v16h __attribute__((ext_vector_type(16)));
typedef float    v8f  __attribute__((ext_vector_type(8)));

#define B_SZ   4096
#define D_IN   784
#define H_SZ   800
#define D_OUT  10
#define KPAD   800     // K padded to multiple of 32
#define T_MAX  512

// workspace layout (bytes), all offsets 256B aligned
#define XF_OFF   0u          // x_fixed f16 [4096][800]   = 6,553,600 B
#define W1H_OFF  6553600u    // W1 hi f16  [800][800]     = 1,280,000 B
#define W1L_OFF  7833600u    // W1 lo f16  [800][800]     = 1,280,000 B
#define CUR1_OFF 9113600u    // cur1 f32   [4096][800]    = 13,107,200 B

// ---- gfx1250 async global->LDS path (guarded; sync staging as fallback) ----
#if defined(__has_builtin)
#if __has_builtin(__builtin_amdgcn_global_load_async_to_lds_b128)
#define USE_ASYNC_LDS 1
#endif
#endif

#ifdef USE_ASYNC_LDS
// Parameter type per hipcc diagnostic: 'int __attribute__((vector_size(16)))
// __device__ *' -> pointer to a 16-byte int vector in address_space(1).
typedef int b128_t __attribute__((vector_size(16)));
typedef __attribute__((address_space(1))) b128_t* gb128_p;
typedef __attribute__((address_space(3))) b128_t* lb128_p;
#if __has_builtin(__builtin_amdgcn_s_wait_asynccnt)
#define STAGE_FENCE() __builtin_amdgcn_s_wait_asynccnt(0)
#else
#define STAGE_FENCE() asm volatile("s_wait_asynccnt 0x0" ::: "memory")
#endif
#else
#define STAGE_FENCE()
#endif

__device__ __forceinline__ unsigned rotl32(unsigned x, int r) {
  return (x << r) | (x >> (32 - r));
}

__device__ __forceinline__ float bits_to_unif(unsigned b) {
  // jax.random.uniform: bitcast((bits >> 9) | 0x3F800000) - 1.0
  return __uint_as_float((b >> 9) | 0x3F800000u) - 1.0f;
}

// ---------------------------------------------------------------------------
// Kernel 1: Bernoulli rate encoding with exact JAX threefry2x32 (key(42)).
// counts = iota(n); halves (t, t+n/2) processed as one threefry pair.
// Also zero-fills the K-padding columns 784..799.
// ---------------------------------------------------------------------------
__global__ __launch_bounds__(256) void snn_encode(const float* __restrict__ x,
                                                  _Float16* __restrict__ xf) {
  const unsigned N  = (unsigned)B_SZ * D_IN;   // 3,211,264 (even)
  const unsigned Hh = N >> 1;                  // 1,605,632
  unsigned t = blockIdx.x * 256u + threadIdx.x;

  if (t < (unsigned)B_SZ * (KPAD - D_IN)) {    // 65,536 pad elements
    unsigned row = t >> 4, col = t & 15u;
    xf[(size_t)row * KPAD + D_IN + col] = (_Float16)0.0f;
  }
  if (t >= Hh) return;

  unsigned x0 = t, x1 = t + Hh;
  const unsigned k0 = 0u, k1 = 42u, k2 = k0 ^ k1 ^ 0x1BD11BDAu;
  x0 += k0; x1 += k1;
#define TF_QR(r) { x0 += x1; x1 = rotl32(x1, r); x1 ^= x0; }
  TF_QR(13) TF_QR(15) TF_QR(26) TF_QR(6)   x0 += k1; x1 += k2 + 1u;
  TF_QR(17) TF_QR(29) TF_QR(16) TF_QR(24)  x0 += k2; x1 += k0 + 2u;
  TF_QR(13) TF_QR(15) TF_QR(26) TF_QR(6)   x0 += k0; x1 += k1 + 3u;
  TF_QR(17) TF_QR(29) TF_QR(16) TF_QR(24)  x0 += k1; x1 += k2 + 4u;
  TF_QR(13) TF_QR(15) TF_QR(26) TF_QR(6)   x0 += k2; x1 += k0 + 5u;
#undef TF_QR
  {
    float u = bits_to_unif(x0);
    unsigned row = t / D_IN, col = t % D_IN;
    xf[(size_t)row * KPAD + col] = (u < x[t]) ? (_Float16)1.0f : (_Float16)0.0f;
  }
  {
    unsigned f = t + Hh;
    float u = bits_to_unif(x1);
    unsigned row = f / D_IN, col = f % D_IN;
    xf[(size_t)row * KPAD + col] = (u < x[f]) ? (_Float16)1.0f : (_Float16)0.0f;
  }
}

// ---------------------------------------------------------------------------
// Kernel 2: split W1 into hi/lo f16 pair (exact products vs {0,1} A-matrix,
// residual captured in f16 subnormals which WMMA preserves). K padded.
// ---------------------------------------------------------------------------
__global__ __launch_bounds__(256) void snn_prep_w1(const float* __restrict__ W1,
                                                   _Float16* __restrict__ w1h,
                                                   _Float16* __restrict__ w1l) {
  unsigned t = blockIdx.x * 256u + threadIdx.x;
  if (t >= (unsigned)H_SZ * KPAD) return;
  unsigned n = t / KPAD, k = t % KPAD;
  float w = (k < D_IN) ? W1[(size_t)n * D_IN + k] : 0.0f;
  _Float16 hi = (_Float16)w;
  _Float16 lo = (_Float16)(w - (float)hi);
  w1h[t] = hi;
  w1l[t] = lo;
}

// ---------------------------------------------------------------------------
// WMMA fragment gather from LDS (padded row stride = 20 dwords = 40 halves).
// 16-bit A-layout (ISA 7.12.2): lanes 0-15 take dwords {0..3, 8..11} of the
// row (K=0..7,16..23); lanes 16-31 take {4..7, 12..15} (K=8..15,24..31).
// B uses the same per-lane pattern with N in place of M (B staged N-major).
// ---------------------------------------------------------------------------
union FragCvt { unsigned u[8]; v16h v; };

__device__ __forceinline__ v16h ld_frag(const unsigned* base, int row, int hiHalf) {
  const unsigned* p = base + row * 20 + (hiHalf ? 4 : 0);
  FragCvt f;
  f.u[0] = p[0];  f.u[1] = p[1];  f.u[2] = p[2];  f.u[3] = p[3];
  f.u[4] = p[8];  f.u[5] = p[9];  f.u[6] = p[10]; f.u[7] = p[11];
  return f.v;
}

// ---------------------------------------------------------------------------
// Stage one 32-wide K chunk of A (128 rows) and B hi/lo (64 rows) into LDS.
// Async path: GLOBAL_LOAD_ASYNC_TO_LDS_B128 (ASYNCcnt, no VGPR round-trip).
// Sync path: b128 loads + ds_store_b128.
// ---------------------------------------------------------------------------
__device__ __forceinline__ void stage_chunk(const _Float16* __restrict__ Xf,
                                            const _Float16* __restrict__ W1h,
                                            const _Float16* __restrict__ W1l,
                                            unsigned* a, unsigned* bh, unsigned* bl,
                                            int bm, int bn, int kk0, int tid) {
#ifdef USE_ASYNC_LDS
#pragma unroll
  for (int i = 0; i < 2; ++i) {
    int idx = tid + 256 * i;
    int r = idx >> 2, s = idx & 3;
    __builtin_amdgcn_global_load_async_to_lds_b128(
        (gb128_p)(Xf + (size_t)(bm + r) * KPAD + kk0 + s * 8),
        (lb128_p)(a + r * 20 + s * 4), 0, 0);
  }
  {
    int r = tid >> 2, s = tid & 3;
    int n = bn + r;
    if (n < H_SZ) {
      __builtin_amdgcn_global_load_async_to_lds_b128(
          (gb128_p)(W1h + (size_t)n * KPAD + kk0 + s * 8),
          (lb128_p)(bh + r * 20 + s * 4), 0, 0);
      __builtin_amdgcn_global_load_async_to_lds_b128(
          (gb128_p)(W1l + (size_t)n * KPAD + kk0 + s * 8),
          (lb128_p)(bl + r * 20 + s * 4), 0, 0);
    }
  }
#else
#pragma unroll
  for (int i = 0; i < 2; ++i) {
    int idx = tid + 256 * i;
    int r = idx >> 2, s = idx & 3;
    uint4 v = *(const uint4*)(Xf + (size_t)(bm + r) * KPAD + kk0 + s * 8);
    *(uint4*)(a + r * 20 + s * 4) = v;
  }
  {
    int r = tid >> 2, s = tid & 3;
    int n = bn + r;
    uint4 vh = make_uint4(0u, 0u, 0u, 0u);
    uint4 vl = make_uint4(0u, 0u, 0u, 0u);
    if (n < H_SZ) {
      vh = *(const uint4*)(W1h + (size_t)n * KPAD + kk0 + s * 8);
      vl = *(const uint4*)(W1l + (size_t)n * KPAD + kk0 + s * 8);
    }
    *(uint4*)(bh + r * 20 + s * 4) = vh;
    *(uint4*)(bl + r * 20 + s * 4) = vl;
  }
#endif
}

// ---------------------------------------------------------------------------
// Kernel 3: cur1 = x_fixed @ W1^T + b1  via v_wmma_f32_16x16x32_f16.
// Block = 256 threads (8 waves), 128x64 output tile, wave = 32x32 (2x2 C
// tiles). K loop: 25 chunks of 32, double-buffered LDS staging (async DMA
// when the toolchain exposes it). 8 WMMA per wave per chunk (hi+lo W1).
// ---------------------------------------------------------------------------
__global__ __launch_bounds__(256) void snn_gemm1(const _Float16* __restrict__ Xf,
                                                 const _Float16* __restrict__ W1h,
                                                 const _Float16* __restrict__ W1l,
                                                 const float* __restrict__ b1,
                                                 float* __restrict__ cur1) {
  __shared__ unsigned lA [2][128 * 20];
  __shared__ unsigned lBh[2][ 64 * 20];
  __shared__ unsigned lBl[2][ 64 * 20];

  const int tid  = threadIdx.x;
  const int lane = tid & 31;
  const int wave = tid >> 5;
  const int wm   = wave >> 1;       // 0..3 -> M sub-tile of 32
  const int wn   = wave & 1;        // 0..1 -> N sub-tile of 32
  const int l15  = lane & 15;
  const int hsel = (lane >> 4) & 1; // lanes 16-31 take the "hi" half pattern
  const int bm   = blockIdx.x * 128;
  const int bn   = blockIdx.y * 64;

  // OOB B rows (n >= 800) are the same for every K chunk: zero them once in
  // both buffers; the staging loop never touches them again.
  if (bn + 64 > H_SZ) {
    int firstPad = H_SZ - bn;
    int total = (64 - firstPad) * 20;
    for (int i = tid; i < total; i += 256) {
      int r = firstPad + i / 20, c = i % 20;
      lBh[0][r * 20 + c] = 0u; lBh[1][r * 20 + c] = 0u;
      lBl[0][r * 20 + c] = 0u; lBl[1][r * 20 + c] = 0u;
    }
  }

  v8f acc[2][2];
#pragma unroll
  for (int i = 0; i < 2; ++i)
#pragma unroll
    for (int j = 0; j < 2; ++j)
#pragma unroll
      for (int e = 0; e < 8; ++e) acc[i][j][e] = 0.0f;

  stage_chunk(Xf, W1h, W1l, &lA[0][0], &lBh[0][0], &lBl[0][0], bm, bn, 0, tid);
  STAGE_FENCE();
  __syncthreads();

  for (int kc = 0; kc < KPAD / 32; ++kc) {
    const int cur = kc & 1;
    const int nxt = cur ^ 1;
    if (kc + 1 < KPAD / 32) {
      stage_chunk(Xf, W1h, W1l, &lA[nxt][0], &lBh[nxt][0], &lBl[nxt][0],
                  bm, bn, (kc + 1) * 32, tid);
    }

    const unsigned* pA  = &lA [cur][0];
    const unsigned* pBh = &lBh[cur][0];
    const unsigned* pBl = &lBl[cur][0];

    v16h a0  = ld_frag(pA,  wm * 32      + l15, hsel);
    v16h a1  = ld_frag(pA,  wm * 32 + 16 + l15, hsel);
    v16h bh0 = ld_frag(pBh, wn * 32      + l15, hsel);
    v16h bh1 = ld_frag(pBh, wn * 32 + 16 + l15, hsel);
    v16h bl0 = ld_frag(pBl, wn * 32      + l15, hsel);
    v16h bl1 = ld_frag(pBl, wn * 32 + 16 + l15, hsel);

    acc[0][0] = __builtin_amdgcn_wmma_f32_16x16x32_f16(false, a0, false, bh0, (short)0, acc[0][0], false, false);
    acc[0][0] = __builtin_amdgcn_wmma_f32_16x16x32_f16(false, a0, false, bl0, (short)0, acc[0][0], false, false);
    acc[0][1] = __builtin_amdgcn_wmma_f32_16x16x32_f16(false, a0, false, bh1, (short)0, acc[0][1], false, false);
    acc[0][1] = __builtin_amdgcn_wmma_f32_16x16x32_f16(false, a0, false, bl1, (short)0, acc[0][1], false, false);
    acc[1][0] = __builtin_amdgcn_wmma_f32_16x16x32_f16(false, a1, false, bh0, (short)0, acc[1][0], false, false);
    acc[1][0] = __builtin_amdgcn_wmma_f32_16x16x32_f16(false, a1, false, bl0, (short)0, acc[1][0], false, false);
    acc[1][1] = __builtin_amdgcn_wmma_f32_16x16x32_f16(false, a1, false, bh1, (short)0, acc[1][1], false, false);
    acc[1][1] = __builtin_amdgcn_wmma_f32_16x16x32_f16(false, a1, false, bl1, (short)0, acc[1][1], false, false);

    STAGE_FENCE();
    __syncthreads();
  }

  // epilogue: C/D layout — VGPR e: lanes 0-15 -> M=e, lanes 16-31 -> M=e+8;
  // N = lane & 15 within the 16-wide tile. Add b1, store f32.
#pragma unroll
  for (int j = 0; j < 2; ++j) {
    int n = bn + wn * 32 + j * 16 + l15;
    if (n < H_SZ) {
      float bv = b1[n];
#pragma unroll
      for (int i = 0; i < 2; ++i) {
        int mb = bm + wm * 32 + i * 16 + hsel * 8;
#pragma unroll
        for (int e = 0; e < 8; ++e) {
          cur1[(size_t)(mb + e) * H_SZ + n] = acc[i][j][e] + bv;
        }
      }
    }
  }
}

// ---------------------------------------------------------------------------
// Kernel 4: fused 32-step IF recurrence. One wave per batch row; v1 state
// lives in registers (25 f32/lane), W2 (10x800, 32KB) + b2 in LDS, cur1 read
// once. cur2 reduced with wave32 xor-butterfly. Output = spike count [B,10].
// T read from device scalar (graph-capture safe).
// ---------------------------------------------------------------------------
__global__ __launch_bounds__(256) void snn_steps(const float* __restrict__ cur1,
                                                 const float* __restrict__ W2,
                                                 const float* __restrict__ b2,
                                                 const int* __restrict__ Tptr,
                                                 float* __restrict__ out) {
  __shared__ float w2s[D_OUT * H_SZ];  // 32,000 B
  __shared__ float b2s[D_OUT];

  const int tid = threadIdx.x;
  for (int i = tid; i < D_OUT * H_SZ; i += 256) w2s[i] = W2[i];
  if (tid < D_OUT) b2s[tid] = b2[tid];
  __syncthreads();

  const int lane = tid & 31;
  const int wave = tid >> 5;
  const int m = blockIdx.x * 8 + wave;  // batch row

  float c1[25], v1[25];
#pragma unroll
  for (int jj = 0; jj < 25; ++jj) {
    c1[jj] = cur1[(size_t)m * H_SZ + jj * 32 + lane];
    v1[jj] = 0.0f;
  }
  float v2[D_OUT], cnt[D_OUT];
#pragma unroll
  for (int j = 0; j < D_OUT; ++j) { v2[j] = 0.0f; cnt[j] = 0.0f; }

  int T = Tptr[0];
  if (T > T_MAX) T = T_MAX;

  for (int t = 0; t < T; ++t) {
    float p[D_OUT];
#pragma unroll
    for (int j = 0; j < D_OUT; ++j) p[j] = 0.0f;

#pragma unroll
    for (int jj = 0; jj < 25; ++jj) {
      float hmem = v1[jj] + c1[jj];
      float s = (hmem >= 1.0f) ? 1.0f : 0.0f;   // heaviside(h1 - V_TH)
      v1[jj] = hmem * (1.0f - s);               // hard reset to 0
      int kk = jj * 32 + lane;
#pragma unroll
      for (int j = 0; j < D_OUT; ++j) p[j] = fmaf(s, w2s[j * H_SZ + kk], p[j]);
    }
#pragma unroll
    for (int j = 0; j < D_OUT; ++j) {
      float v = p[j];
#pragma unroll
      for (int off = 16; off >= 1; off >>= 1) v += __shfl_xor(v, off, 32);
      float cur2 = v + b2s[j];
      float h2 = v2[j] + cur2;
      float s2 = (h2 >= 1.0f) ? 1.0f : 0.0f;
      v2[j] = h2 * (1.0f - s2);
      cnt[j] += s2;
    }
  }

#pragma unroll
  for (int j = 0; j < D_OUT; ++j) {
    if (lane == j) out[(size_t)m * D_OUT + j] = cnt[j];
  }
}

// ---------------------------------------------------------------------------
extern "C" void kernel_launch(void* const* d_in, const int* in_sizes, int n_in,
                              void* d_out, int out_size, void* d_ws, size_t ws_size,
                              hipStream_t stream) {
  (void)in_sizes; (void)n_in; (void)out_size; (void)ws_size;
  const float* x  = (const float*)d_in[0];
  const float* W1 = (const float*)d_in[1];
  const float* b1 = (const float*)d_in[2];
  const float* W2 = (const float*)d_in[3];
  const float* b2 = (const float*)d_in[4];
  const int*   T  = (const int*)d_in[5];
  float* out = (float*)d_out;

  char* ws = (char*)d_ws;
  _Float16* xf   = (_Float16*)(ws + XF_OFF);
  _Float16* w1h  = (_Float16*)(ws + W1H_OFF);
  _Float16* w1l  = (_Float16*)(ws + W1L_OFF);
  float*    cur1 = (float*)   (ws + CUR1_OFF);

  snn_encode <<<6272, 256, 0, stream>>>(x, xf);                  // 1,605,632 / 256
  snn_prep_w1<<<2500, 256, 0, stream>>>(W1, w1h, w1l);           // 640,000 / 256
  snn_gemm1  <<<dim3(32, 13), 256, 0, stream>>>(xf, w1h, w1l, b1, cur1);
  snn_steps  <<<512, 256, 0, stream>>>(cur1, W2, b2, T, out);
}
